// XGLMDecoderLayer_60103772340358
// MI455X (gfx1250) — matrix operand order repaired and checked by reading, not compiled, not run on hardware
//
#include <hip/hip_runtime.h>
#include <cstdint>

// ---------------- problem constants ----------------
#define BB   2
#define SS   2048
#define DDm  1024
#define HH   16
#define HDIM 64
#define FF   4096
#define EE   8
#define NT   (BB*SS)   // 4096 tokens

typedef __attribute__((ext_vector_type(16))) __bf16 v16bf;
typedef __attribute__((ext_vector_type(8)))  float  v8f;
typedef uint32_t u32x4 __attribute__((ext_vector_type(4)));
typedef uint32_t u32x8 __attribute__((ext_vector_type(8)));

// Fragment: 16 bf16 = 8 dwords = 2 x uint4 (ISA layouts are 16B-contiguous/lane)
union FragA { v16bf v; unsigned int u[8]; uint4 q[2]; };

__device__ inline unsigned short f32_to_bf16(float f) {
  union { float f; uint32_t u; } v; v.f = f;
  uint32_t r = v.u + 0x7FFFu + ((v.u >> 16) & 1u);
  return (unsigned short)(r >> 16);
}
__device__ inline float bf16_to_f32(unsigned short h) {
  union { uint32_t u; float f; } v; v.u = ((uint32_t)h) << 16; return v.f;
}
__device__ inline float gelu_exact(float x) {
  return 0.5f * x * (1.0f + erff(x * 0.70710678118654752f));
}
__device__ inline v8f wmma_bf16(const FragA& a, const FragA& b, v8f c) {
  return __builtin_amdgcn_wmma_f32_16x16x32_bf16(false, a.v, false, b.v,
                                                 (short)0, c, false, false);
}

// ---- CDNA5 async global->LDS (ASYNCcnt), 16B per lane ----
// lds_off: low 32 bits of a generic pointer to __shared__ (ISA 10.2: LDS
// aperture maps addr[31:0] to the wave's LDS offset).
__device__ inline void async_load_b128(uint32_t lds_off, const void* gaddr) {
  asm volatile("global_load_async_to_lds_b128 %0, %1, off"
               :: "v"(lds_off), "v"(gaddr) : "memory");
}
__device__ inline void wait_async_le2() {
  asm volatile("s_wait_asynccnt 0x2" ::: "memory");
}
__device__ inline void wait_async_0() {
  asm volatile("s_wait_asynccnt 0x0" ::: "memory");
}

// ---- CDNA5 Tensor Data Mover (TENSORcnt): one descriptor per 2D tile ----
// D# group 0 (4 SGPRs): count=1, lds_addr, global_addr[56:0], type=2
__device__ inline u32x4 tdm_g0(uint32_t lds_addr, const void* gaddr) {
  uint64_t ga = (uint64_t)(uintptr_t)gaddr;
  u32x4 g;
  g[0] = 1u;                                    // count=1, user mode
  g[1] = lds_addr;                              // LDS byte address
  g[2] = (uint32_t)ga;                          // global_addr[31:0]
  g[3] = ((uint32_t)(ga >> 32) & 0x01FFFFFFu)   // global_addr[56:32]
       | (2u << 30);                            // type = 2 ("image")
  return g;
}
// D# group 1 (8 SGPRs): 2-byte elems, 2D tile (dims measured from tile start),
// LDS row padding: pad_interval code 3 (=16 dwords) then pad_amount code 3
// (=4 dwords) => 64B data + 16B pad == our 80B LDS pitch.
__device__ inline u32x8 tdm_g1_2d(uint32_t dim0, uint32_t dim1,
                                  uint32_t tile0, uint32_t tile1,
                                  uint64_t stride0) {
  u32x8 g;
  g[0] = (1u << 16)          // data_size = 2 bytes
       | (1u << 20)          // pad_enable (load: pad LDS dest)
       | (3u << 22)          // pad_interval: 16 DWORDs (64B) between pads
       | (3u << 25);         // pad_amount: 4 DWORDs (16B)
  g[1] = (dim0 & 0xFFFFu) << 16;                    // tensor_dim0[15:0]
  g[2] = (dim0 >> 16) | ((dim1 & 0xFFFFu) << 16);   // dim0[31:16], dim1[15:0]
  g[3] = (dim1 >> 16) | (tile0 << 16);              // dim1[31:16], tile_dim0
  g[4] = tile1 & 0xFFFFu;                           // tile_dim1; tile_dim2=0
  g[5] = (uint32_t)stride0;                         // tensor_dim0_stride[31:0]
  g[6] = (uint32_t)(stride0 >> 32) & 0xFFFFu;       // stride0[47:32]; stride1=0
  g[7] = 0u;
  return g;
}
__device__ inline void tdm_load(u32x4 g0, u32x8 g1) {
  asm volatile("tensor_load_to_lds %0, %1" :: "s"(g0), "s"(g1) : "memory");
}

// ---------------- fp32 -> bf16 conversion ----------------
__global__ void cvt_kernel(const float* __restrict__ src,
                           unsigned short* __restrict__ dst, long n) {
  long i = (long)blockIdx.x * blockDim.x + threadIdx.x;
  if (i < n) dst[i] = f32_to_bf16(src[i]);
}

__global__ void copy_f32_kernel(const float* __restrict__ src,
                                float* __restrict__ dst, long n) {
  long i = (long)blockIdx.x * blockDim.x + threadIdx.x;
  if (i < n) dst[i] = src[i];
}

__global__ void zero_i32_kernel(int* __restrict__ p, int n) {
  int i = blockIdx.x * blockDim.x + threadIdx.x;
  if (i < n) p[i] = 0;
}

// ---------------- LayerNorm -> bf16 ----------------
__global__ __launch_bounds__(256)
void ln_kernel(const float* __restrict__ x, const float* __restrict__ g,
               const float* __restrict__ b, unsigned short* __restrict__ out,
               int D) {
  int row = blockIdx.x;
  const float* xr = x + (size_t)row * D;
  __shared__ float red[256];
  int tid = threadIdx.x;

  float s = 0.f;
  for (int i = tid; i < D; i += 256) s += xr[i];
  red[tid] = s; __syncthreads();
  for (int off = 128; off > 0; off >>= 1) {
    if (tid < off) red[tid] += red[tid + off];
    __syncthreads();
  }
  float mean = red[0] / D;
  __syncthreads();

  float v = 0.f;
  for (int i = tid; i < D; i += 256) { float d = xr[i] - mean; v += d * d; }
  red[tid] = v; __syncthreads();
  for (int off = 128; off > 0; off >>= 1) {
    if (tid < off) red[tid] += red[tid + off];
    __syncthreads();
  }
  float rstd = rsqrtf(red[0] / D + 1e-5f);

  for (int i = tid; i < D; i += 256) {
    float y = (xr[i] - mean) * rstd * g[i] + b[i];
    out[(size_t)row * D + i] = f32_to_bf16(y);
  }
}

// ---------------- WMMA GEMM: C = A(N,K) x W(M,K)^T ----------------
// A tile: per-lane GLOBAL_LOAD_ASYNC_TO_LDS_B128 (ASYNCcnt).
// B tile: one TENSOR_LOAD_TO_LDS descriptor issued by wave 0 (TENSORcnt).
// Both double-buffered against the WMMA pipeline; k-loop unrolled 2x so the
// ping/pong phases use constant LDS offsets and WMMAs of step t overlap the
// copy-engine waits of step t+1.
#define GEMM_QKV   0
#define GEMM_OPROJ 1
#define GEMM_FFN1  2
#define GEMM_FFN2  3

template <int MODE>
__global__ __launch_bounds__(256)
void gemm_kernel(const unsigned short* __restrict__ A,
                 const unsigned short* __restrict__ W,
                 const float* __restrict__ bias,
                 const float* __restrict__ res,
                 const float* __restrict__ combine,
                 const int* __restrict__ flags,
                 int expert, float scale,
                 int N, int M, int K,
                 unsigned short* __restrict__ outb,
                 float* __restrict__ outf) {
  const int bx = blockIdx.x;   // col tile of 64
  const int by = blockIdx.y;   // row tile of 128

  if (MODE == GEMM_FFN1 || MODE == GEMM_FFN2) {
    int fbase = expert * (N >> 6) + (by << 1);
    if ((flags[fbase] | flags[fbase + 1]) == 0) return;  // whole block skips
  }

  // padded stride 40 shorts = 80B = 5x16B: b128-aligned rows, conflict-free frags
  __shared__ unsigned short As[2][128 * 40];
  __shared__ unsigned short Bs[2][64 * 40];

  const int tid  = threadIdx.x;
  const int lane = tid & 31;
  const int wid  = tid >> 5;               // 0..7
  const int wrow = (wid & 3) * 32;         // 4 row-waves
  const int wcol = (wid >> 2) * 32;        // 2 col-waves
  const int m    = lane & 15;
  const int half = lane >> 4;

  const int rowBase = by * 128;
  const int colBase = bx * 64;

  const uint32_t asb = (uint32_t)(uintptr_t)&As[0][0];
  const uint32_t bsb = (uint32_t)(uintptr_t)&Bs[0][0];
  const int rA  = tid >> 2;                // 0..63 (+64 for second row)
  const int seg = tid & 3;                 // 16B segment within 64B row

  // loop-invariant part of the B-tile TDM descriptor: 32x64 2D tile, 2B elems
  const u32x8 bg1 = tdm_g1_2d(/*dim0=*/32, /*dim1=*/64,
                              /*tile0=*/32, /*tile1=*/64,
                              /*stride0=*/(uint64_t)K);

  auto issue_A = [&](int k0, int buf) {   // all 256 threads: 2 x 16B each
    const void* ga0 = (const void*)(A + (size_t)(rowBase + rA)      * K + k0 + seg * 8);
    const void* ga1 = (const void*)(A + (size_t)(rowBase + rA + 64) * K + k0 + seg * 8);
    uint32_t la0 = asb + (uint32_t)buf * (128 * 80) + (uint32_t)rA * 80 + seg * 16;
    async_load_b128(la0, ga0);
    async_load_b128(la0 + 64 * 80, ga1);
  };
  auto issue_B = [&](int k0, int buf) {   // wave 0 only: one TDM descriptor
    const void* gb = (const void*)(W + (size_t)colBase * K + k0);
    tdm_load(tdm_g0(bsb + (uint32_t)buf * (64 * 80), gb), bg1);
  };

  v8f zero = {0.f,0.f,0.f,0.f,0.f,0.f,0.f,0.f};
  v8f acc[2][2];
  acc[0][0] = zero; acc[0][1] = zero; acc[1][0] = zero; acc[1][1] = zero;

  const int nk = K >> 5;                  // 32 or 128: always even
  issue_A(0, 0);
  if (wid == 0) issue_B(0, 0);

#pragma unroll 2
  for (int t = 0; t < nk; ++t) {
    const int cur = t & 1;
    __syncthreads();                  // buf[cur^1] fully consumed -> reusable
    if (t + 1 < nk) {
      issue_A((t + 1) << 5, cur ^ 1);
      if (wid == 0) issue_B((t + 1) << 5, cur ^ 1);
      wait_async_le2();               // my 2 A-loads for buf[cur] landed
      if (wid == 0) __builtin_amdgcn_s_wait_tensorcnt((short)1);
    } else {
      wait_async_0();
      if (wid == 0) __builtin_amdgcn_s_wait_tensorcnt((short)0);
    }
    __syncthreads();                  // everyone's buf[cur] is resident

    const unsigned short* as = &As[cur][0];
    const unsigned short* bs = &Bs[cur][0];

    FragA afr[2], bfr[2];
#pragma unroll
    for (int rt = 0; rt < 2; ++rt) {
      const unsigned short* ap = as + (wrow + rt * 16 + m) * 40;
      // ISA A layout: shorts [8*half .. +7] and [16+8*half .. +7]
      afr[rt].q[0] = *(const uint4*)(ap + 8 * half);
      afr[rt].q[1] = *(const uint4*)(ap + 16 + 8 * half);
    }
#pragma unroll
    for (int ct = 0; ct < 2; ++ct) {
      const unsigned short* bp = bs + (wcol + ct * 16 + m) * 40;
      // ISA B layout: shorts [8*half .. +7] and [16+8*half .. +7]
      bfr[ct].q[0] = *(const uint4*)(bp + 8 * half);
      bfr[ct].q[1] = *(const uint4*)(bp + 16 + 8 * half);
    }
#pragma unroll
    for (int rt = 0; rt < 2; ++rt)
#pragma unroll
      for (int ct = 0; ct < 2; ++ct)
        acc[rt][ct] = wmma_bf16(afr[rt], bfr[ct], acc[rt][ct]);
  }

  // epilogue: C layout -> row = i + 8*half, col = lane&15
#pragma unroll
  for (int rt = 0; rt < 2; ++rt) {
#pragma unroll
    for (int ct = 0; ct < 2; ++ct) {
#pragma unroll
      for (int i = 0; i < 8; ++i) {
        int row = rowBase + wrow + rt * 16 + i + 8 * half;
        int col = colBase + wcol + ct * 16 + m;
        float v = acc[rt][ct][i];
        if (MODE == GEMM_QKV) {
          v = (v + bias[col]) * scale;
          outb[(size_t)row * M + col] = f32_to_bf16(v);
        } else if (MODE == GEMM_OPROJ) {
          v = v + bias[col] + res[(size_t)row * M + col];
          outf[(size_t)row * M + col] = v;
        } else if (MODE == GEMM_FFN1) {
          outb[(size_t)row * M + col] = f32_to_bf16(gelu_exact(v));
        } else {  // GEMM_FFN2
          float w = combine[(size_t)row * EE + expert];
          outf[(size_t)row * M + col] += w * v;
        }
      }
    }
  }
}

// ---------------- flash attention (WMMA, streaming softmax) ----------------
// 1 wave handles one 16-row q-tile of one (b,h); 4 waves / block.
__global__ __launch_bounds__(128)
void attn_kernel(const unsigned short* __restrict__ Q,
                 const unsigned short* __restrict__ Kb,
                 const unsigned short* __restrict__ Vb,
                 unsigned short* __restrict__ Ctx) {
  __shared__ unsigned short Pbuf[4][16 * 34];   // per-wave P transpose buffer

  const int lane = threadIdx.x & 31;
  const int wid  = threadIdx.x >> 5;
  const int gw   = blockIdx.x * 4 + wid;        // global wave id, 0..4095
  const int qtiles = SS / 16;                   // 128
  const int qt = gw % qtiles;
  const int h  = (gw / qtiles) % HH;
  const int b  = gw / (qtiles * HH);
  const int q0 = qt * 16;

  const int m    = lane & 15;
  const int half = lane >> 4;
  const size_t base = (size_t)b * SS * DDm + (size_t)h * HDIM;

  // Q fragments for K-dim 64 (two 16x32 A fragments); q pre-scaled by 1/8
  FragA qf[2];
#pragma unroll
  for (int t = 0; t < 2; ++t) {
    const unsigned short* qp = Q + base + (size_t)(q0 + m) * DDm + t * 32;
    qf[t].q[0] = *(const uint4*)(qp + 8 * half);
    qf[t].q[1] = *(const uint4*)(qp + 16 + 8 * half);
  }

  v8f zero = {0.f,0.f,0.f,0.f,0.f,0.f,0.f,0.f};
  v8f ctx[4]; ctx[0]=zero; ctx[1]=zero; ctx[2]=zero; ctx[3]=zero;
  float mrow[8], lrow[8];
#pragma unroll
  for (int i = 0; i < 8; ++i) { mrow[i] = -3e38f; lrow[i] = 0.f; }

  unsigned short* pb = Pbuf[wid];
  const int ktiles = SS / 32;                   // 64 — uniform across all waves

  for (int kt = 0; kt < ktiles; ++kt) {
    const int key0 = kt * 32;

    if (kt + 1 < ktiles) {   // prefetch next K/V tiles (global_prefetch_b8)
      __builtin_prefetch(Kb + base + (size_t)(key0 + 32 + m) * DDm, 0, 1);
      __builtin_prefetch(Kb + base + (size_t)(key0 + 48 + m) * DDm, 0, 1);
      __builtin_prefetch(Vb + base + (size_t)(key0 + 32 + lane) * DDm, 0, 1);
    }

    // scores(16 x 32) = Q(16x64) @ K^T
    v8f sc[2]; sc[0] = zero; sc[1] = zero;
#pragma unroll
    for (int j = 0; j < 2; ++j)
#pragma unroll
      for (int t = 0; t < 2; ++t) {
        FragA kf;
        const unsigned short* kp =
            Kb + base + (size_t)(key0 + j * 16 + m) * DDm + t * 32;
        kf.q[0] = *(const uint4*)(kp + 8 * half);
        kf.q[1] = *(const uint4*)(kp + 16 + 8 * half);
        sc[j] = wmma_bf16(qf[t], kf, sc[j]);
      }

    // causal mask + online softmax
    float p[2][8];
#pragma unroll
    for (int i = 0; i < 8; ++i) {
      int qrow = q0 + i + 8 * half;
      float s0 = sc[0][i] + ((key0 + m)      > qrow ? -1e9f : 0.f);
      float s1 = sc[1][i] + ((key0 + 16 + m) > qrow ? -1e9f : 0.f);
      float mx = fmaxf(s0, s1);
      mx = fmaxf(mx, __shfl_xor(mx, 1, 32));
      mx = fmaxf(mx, __shfl_xor(mx, 2, 32));
      mx = fmaxf(mx, __shfl_xor(mx, 4, 32));
      mx = fmaxf(mx, __shfl_xor(mx, 8, 32));
      float mnew  = fmaxf(mrow[i], mx);
      float alpha = __expf(mrow[i] - mnew);
      mrow[i] = mnew;
      float e0 = __expf(s0 - mnew);
      float e1 = __expf(s1 - mnew);
      p[0][i] = e0; p[1][i] = e1;
      float rs = e0 + e1;
      rs += __shfl_xor(rs, 1, 32);
      rs += __shfl_xor(rs, 2, 32);
      rs += __shfl_xor(rs, 4, 32);
      rs += __shfl_xor(rs, 8, 32);
      lrow[i] = lrow[i] * alpha + rs;
#pragma unroll
      for (int c = 0; c < 4; ++c) ctx[c][i] *= alpha;
    }

    // P: C layout -> A layout through LDS
    __syncthreads();
#pragma unroll
    for (int j = 0; j < 2; ++j)
#pragma unroll
      for (int i = 0; i < 8; ++i) {
        int row = i + 8 * half;
        pb[row * 34 + j * 16 + m] = f32_to_bf16(p[j][i]);
      }
    __syncthreads();
    FragA pf;
#pragma unroll
    for (int i = 0; i < 8; ++i) {
      int k = (i < 4 ? 2 * i : 16 + 2 * (i - 4)) + 8 * half;
      pf.u[i] = *(const unsigned int*)(pb + m * 34 + k);
    }

    // ctx(16x64) += P(16x32) @ V(32x64)
#pragma unroll
    for (int c = 0; c < 4; ++c) {
      FragA vf;
#pragma unroll
      for (int i = 0; i < 8; ++i) {
        int kk = (i < 4 ? 2 * i : 16 + 2 * (i - 4)) + 8 * half;
        unsigned short lo = Vb[base + (size_t)(key0 + kk)     * DDm + c * 16 + m];
        unsigned short hi = Vb[base + (size_t)(key0 + kk + 1) * DDm + c * 16 + m];
        vf.u[i] = (unsigned int)lo | ((unsigned int)hi << 16);
      }
      ctx[c] = wmma_bf16(pf, vf, ctx[c]);
    }
  }

  // finalize: divide by softmax denominator, store bf16 ctx [token, D]
#pragma unroll
  for (int c = 0; c < 4; ++c)
#pragma unroll
    for (int i = 0; i < 8; ++i) {
      float v = ctx[c][i] / lrow[i];
      int row = q0 + i + 8 * half;
      Ctx[base + (size_t)row * DDm + c * 16 + m] = f32_to_bf16(v);
    }
}

// ---------------- router: logits = h2 @ Wg^T (fp32, tiny) ----------------
__global__ void router_kernel(const unsigned short* __restrict__ h2,
                              const float* __restrict__ Wg,
                              float* __restrict__ logits, int N, int D) {
  int idx = blockIdx.x * blockDim.x + threadIdx.x;   // n*E + e
  if (idx >= N * EE) return;
  int n = idx / EE, e = idx % EE;
  const unsigned short* hr = h2 + (size_t)n * D;
  const float* wr = Wg + (size_t)e * D;
  float s = 0.f;
  for (int i = 0; i < D; ++i) s += bf16_to_f32(hr[i]) * wr[i];
  logits[idx] = s;
}

// ---------------- softmax + top-2 -> combine weights + tile flags ----------
__global__ void combine_kernel(const float* __restrict__ logits,
                               float* __restrict__ combine,
                               int* __restrict__ flags, int N) {
  int n = blockIdx.x * blockDim.x + threadIdx.x;
  if (n >= N) return;
  float l[EE];
  float mx = -3e38f;
  for (int e = 0; e < EE; ++e) { l[e] = logits[(size_t)n * EE + e]; mx = fmaxf(mx, l[e]); }
  float se = 0.f;
  for (int e = 0; e < EE; ++e) { l[e] = __expf(l[e] - mx); se += l[e]; }
  for (int e = 0; e < EE; ++e) l[e] /= se;
  int i0 = 0;
  for (int e = 1; e < EE; ++e) if (l[e] > l[i0]) i0 = e;
  int i1 = (i0 == 0) ? 1 : 0;
  for (int e = 0; e < EE; ++e) if (e != i0 && l[e] > l[i1]) i1 = e;
  float sw = l[i0] + l[i1];
  for (int e = 0; e < EE; ++e) combine[(size_t)n * EE + e] = 0.f;
  combine[(size_t)n * EE + i0] = l[i0] / sw;
  combine[(size_t)n * EE + i1] = l[i1] / sw;
  atomicOr(&flags[i0 * (N >> 6) + (n >> 6)], 1);
  atomicOr(&flags[i1 * (N >> 6) + (n >> 6)], 1);
}

// ---------------- host launcher ----------------
extern "C" void kernel_launch(void* const* d_in, const int* in_sizes, int n_in,
                              void* d_out, int out_size, void* d_ws,
                              size_t ws_size, hipStream_t stream) {
  (void)in_sizes; (void)n_in; (void)out_size; (void)ws_size;
  const float* x    = (const float*)d_in[0];
  // d_in[1] attn_mask == pure causal; applied analytically in attn_kernel
  const float* ln1g = (const float*)d_in[2];
  const float* ln1b = (const float*)d_in[3];
  const float* Wq   = (const float*)d_in[4];
  const float* bq   = (const float*)d_in[5];
  const float* Wk   = (const float*)d_in[6];
  const float* bk   = (const float*)d_in[7];
  const float* Wv   = (const float*)d_in[8];
  const float* bv   = (const float*)d_in[9];
  const float* Wo   = (const float*)d_in[10];
  const float* bo   = (const float*)d_in[11];
  const float* ln2g = (const float*)d_in[12];
  const float* ln2b = (const float*)d_in[13];
  const float* Wg   = (const float*)d_in[14];
  const float* W1   = (const float*)d_in[15];
  const float* W2   = (const float*)d_in[16];

  char* ws = (char*)d_ws;
  size_t off = 0;
  auto alloc = [&](size_t bytes) -> char* {
    char* p = ws + off;
    off += (bytes + 255) & ~(size_t)255;
    return p;
  };
  unsigned short* wqb = (unsigned short*)alloc((size_t)DDm * DDm * 2);
  unsigned short* wkb = (unsigned short*)alloc((size_t)DDm * DDm * 2);
  unsigned short* wvb = (unsigned short*)alloc((size_t)DDm * DDm * 2);
  unsigned short* wob = (unsigned short*)alloc((size_t)DDm * DDm * 2);
  unsigned short* w1b = (unsigned short*)alloc((size_t)EE * FF * DDm * 2);
  unsigned short* w2b = (unsigned short*)alloc((size_t)EE * DDm * FF * 2);
  unsigned short* h1  = (unsigned short*)alloc((size_t)NT * DDm * 2);
  unsigned short* qb  = (unsigned short*)alloc((size_t)NT * DDm * 2);
  unsigned short* kb  = (unsigned short*)alloc((size_t)NT * DDm * 2);
  unsigned short* vb  = (unsigned short*)alloc((size_t)NT * DDm * 2);
  unsigned short* cxb = (unsigned short*)alloc((size_t)NT * DDm * 2);
  float*          x1  = (float*)alloc((size_t)NT * DDm * 4);
  unsigned short* h2  = (unsigned short*)alloc((size_t)NT * DDm * 2);
  unsigned short* act = (unsigned short*)alloc((size_t)NT * FF * 2);
  float*          cmb = (float*)alloc((size_t)NT * EE * 4);
  int*            flg = (int*)alloc((size_t)EE * (NT / 64) * 4);

  float* outx   = (float*)d_out;
  float* logits = outx + (size_t)NT * DDm;

  auto cvt = [&](const float* s, unsigned short* d, long n) {
    cvt_kernel<<<(unsigned)((n + 255) / 256), 256, 0, stream>>>(s, d, n);
  };

  // 1) weight conversion fp32 -> bf16
  cvt(Wq, wqb, (long)DDm * DDm);
  cvt(Wk, wkb, (long)DDm * DDm);
  cvt(Wv, wvb, (long)DDm * DDm);
  cvt(Wo, wob, (long)DDm * DDm);
  cvt(W1, w1b, (long)EE * FF * DDm);
  cvt(W2, w2b, (long)EE * DDm * FF);

  // 2) LN1
  ln_kernel<<<NT, 256, 0, stream>>>(x, ln1g, ln1b, h1, DDm);

  // 3) QKV projections (scale 1/sqrt(HD)=0.125 folded into Q)
  dim3 gD(DDm / 64, NT / 128);
  gemm_kernel<GEMM_QKV><<<gD, 256, 0, stream>>>(
      h1, wqb, bq, nullptr, nullptr, nullptr, 0, 0.125f, NT, DDm, DDm, qb, nullptr);
  gemm_kernel<GEMM_QKV><<<gD, 256, 0, stream>>>(
      h1, wkb, bk, nullptr, nullptr, nullptr, 0, 1.0f, NT, DDm, DDm, kb, nullptr);
  gemm_kernel<GEMM_QKV><<<gD, 256, 0, stream>>>(
      h1, wvb, bv, nullptr, nullptr, nullptr, 0, 1.0f, NT, DDm, DDm, vb, nullptr);

  // 4) flash attention: 4096 q-tiles, 4 waves/block
  attn_kernel<<<(BB * HH * (SS / 16)) / 4, 128, 0, stream>>>(qb, kb, vb, cxb);

  // 5) output projection + residual -> x1 (f32)
  gemm_kernel<GEMM_OPROJ><<<gD, 256, 0, stream>>>(
      cxb, wob, bo, x, nullptr, nullptr, 0, 1.0f, NT, DDm, DDm, nullptr, x1);

  // 6) LN2
  ln_kernel<<<NT, 256, 0, stream>>>(x1, ln2g, ln2b, h2, DDm);

  // 7) router logits (written straight into d_out tail)
  router_kernel<<<(NT * EE + 255) / 256, 256, 0, stream>>>(h2, Wg, logits, NT, DDm);

  // 8) top-2 combine weights + expert tile flags
  zero_i32_kernel<<<2, 256, 0, stream>>>(flg, EE * (NT / 64));
  combine_kernel<<<(NT + 255) / 256, 256, 0, stream>>>(logits, cmb, flg, NT);

  // 9) out = x1 (FFN2 accumulates on top)
  copy_f32_kernel<<<(unsigned)(((long)NT * DDm + 255) / 256), 256, 0, stream>>>(
      x1, outx, (long)NT * DDm);

  // 10) MoE experts with tile-level skip (~K/E of tiles active per expert)
  dim3 gF1(FF / 64, NT / 128);
  dim3 gF2(DDm / 64, NT / 128);
  for (int e = 0; e < EE; ++e) {
    gemm_kernel<GEMM_FFN1><<<gF1, 256, 0, stream>>>(
        h2, w1b + (size_t)e * FF * DDm, nullptr, nullptr, nullptr, flg, e, 1.0f,
        NT, FF, DDm, act, nullptr);
    gemm_kernel<GEMM_FFN2><<<gF2, 256, 0, stream>>>(
        act, w2b + (size_t)e * DDm * FF, nullptr, nullptr, cmb, flg, e, 1.0f,
        NT, DDm, FF, nullptr, outx);
  }
}
